// Memory_28578712388135
// MI455X (gfx1250) — compile-verified, matrix operand order checked by reading, not run on hardware
//
#include <hip/hip_runtime.h>
#include <stddef.h>

// ---------------------------------------------------------------------------
// CDNA5 (gfx1250) implementation: bf16 WMMA (v_wmma_f32_16x16x32_bf16) for all
// GEMMs, fp32 accumulation, wave32 cross-lane softmax / LayerNorm, and
// GLOBAL_LOAD_ASYNC_TO_LDS_B128 (ASYNCcnt) staging of GEMM tiles when the
// toolchain exposes the gfx1250 async builtins.
// ---------------------------------------------------------------------------

typedef __bf16 bf16x16 __attribute__((ext_vector_type(16)));
typedef float  f32x8   __attribute__((ext_vector_type(8)));
typedef int    i32x4   __attribute__((ext_vector_type(4)));
typedef unsigned short u16;

#define NTOK   32768   // B*S = 64*512
#define CDIM   512
#define NHEAD  8
#define HDIM   64
#define NSLOT  112

#if defined(__HIP_DEVICE_COMPILE__) && \
    __has_builtin(__builtin_amdgcn_global_load_async_to_lds_b128) && \
    __has_builtin(__builtin_amdgcn_s_wait_asynccnt)
#define ASYNC_LDS 1
#else
#define ASYNC_LDS 0
#endif

template<typename T> struct IsFloat { static constexpr bool value = false; };
template<>           struct IsFloat<float> { static constexpr bool value = true; };

union FragB16 { bf16x16 v; uint4 q[2]; };

__device__ inline u16 f2bf(float x) {
    unsigned u = __float_as_uint(x);
    u += 0x7fffu + ((u >> 16) & 1u);            // round-to-nearest-even
    return (u16)(u >> 16);
}

__device__ inline f32x8 zero8() {
    f32x8 z;
#pragma unroll
    for (int i = 0; i < 8; ++i) z[i] = 0.f;
    return z;
}

__device__ inline f32x8 wmma_bf16(const FragB16& a, const FragB16& b, f32x8 c) {
    return __builtin_amdgcn_wmma_f32_16x16x32_bf16(
        false, a.v, false, b.v, (short)0, c, false, false);
}

#if ASYNC_LDS
// 16-byte async DMA: global -> LDS, no VGPR round-trip, tracked by ASYNCcnt.
__device__ inline void async_copy_b128(const void* g, void* l) {
    __builtin_amdgcn_global_load_async_to_lds_b128(
        (__attribute__((address_space(1))) i32x4*)g,
        (__attribute__((address_space(3))) i32x4*)l,
        0, 0);
}
__device__ inline void async_wait_all() { __builtin_amdgcn_s_wait_asynccnt(0); }
#else
__device__ inline void async_wait_all() {}
#endif

// full-wave (32-lane) reductions
__device__ inline float wsum32(float v) {
#pragma unroll
    for (int m = 16; m; m >>= 1) v += __shfl_xor(v, m, 32);
    return v;
}
// 16-lane-group reductions (rows of a WMMA D tile live in half-wave groups)
__device__ inline float wsum16(float v) {
#pragma unroll
    for (int m = 8; m; m >>= 1) v += __shfl_xor(v, m, 32);
    return v;
}
__device__ inline float wmax16(float v) {
#pragma unroll
    for (int m = 8; m; m >>= 1) v = fmaxf(v, __shfl_xor(v, m, 32));
    return v;
}

// ---------------------------------------------------------------------------
// Generic tiled GEMM:  C[M,N] = A[M,K] @ B'[N,K]ᵀ (+bias).  A fp32-or-bf16,
// B stored N-major [Ncols][K] bf16 so both tiles are straight block copies
// (async-to-LDS capable).  Block tile 128x64, BK=32, 8 waves (4x2), wave tile
// 32x32 (2x2 WMMA).  Double-buffered LDS.
// ---------------------------------------------------------------------------
template<typename AT>
__global__ __launch_bounds__(256) void gemm_bias_kernel(
    const AT* __restrict__ A, const u16* __restrict__ B,
    const float* __restrict__ bias, float* __restrict__ C,
    int M, int Ncols, int K)
{
    __shared__ u16 sA[2][128][32];
    __shared__ u16 sB[2][64][32];

    const int tid  = threadIdx.x;
    const int lane = tid & 31, w = tid >> 5;
    const int l16  = lane & 15, half = lane >> 4;
    const int wm   = (w & 3) * 32, wn = (w >> 2) * 32;
    const int bm   = blockIdx.y * 128;
    const int n0   = blockIdx.x * 64;

    f32x8 acc[2][2];
#pragma unroll
    for (int i = 0; i < 2; ++i)
#pragma unroll
        for (int j = 0; j < 2; ++j) acc[i][j] = zero8();

    auto loadTiles = [&](int kt, int buf) {
        const int k0 = kt * 32;
        if constexpr (IsFloat<AT>::value) {
            // fp32 source: load, convert to bf16, store to LDS (DScnt path)
#pragma unroll
            for (int i = 0; i < 4; ++i) {                    // 1024 float4 slots
                int idx = i * 256 + tid;
                int row = idx >> 3, kv = (idx & 7) * 4;
                float4 f = *reinterpret_cast<const float4*>(
                    &A[(size_t)(bm + row) * K + k0 + kv]);
                uint2 p;
                p.x = (unsigned)f2bf(f.x) | ((unsigned)f2bf(f.y) << 16);
                p.y = (unsigned)f2bf(f.z) | ((unsigned)f2bf(f.w) << 16);
                *reinterpret_cast<uint2*>(&sA[buf][row][kv]) = p;
            }
        } else {
            // bf16 source: pure block copy -> async DMA to LDS
#pragma unroll
            for (int i = 0; i < 2; ++i) {                    // 512 uint4 slots
                int idx = i * 256 + tid;
                int row = idx >> 2, kv = (idx & 3) * 8;
                const void* g = &A[(size_t)(bm + row) * K + k0 + kv];
#if ASYNC_LDS
                async_copy_b128(g, &sA[buf][row][kv]);
#else
                *reinterpret_cast<uint4*>(&sA[buf][row][kv]) =
                    *reinterpret_cast<const uint4*>(g);
#endif
            }
        }
        {   // B tile: 64 rows (n) x 32 (k), N-major source -> block copy
            int row = tid >> 2, kv = (tid & 3) * 8;          // 256 uint4 slots
            const void* g = &B[(size_t)(n0 + row) * K + k0 + kv];
#if ASYNC_LDS
            async_copy_b128(g, &sB[buf][row][kv]);
#else
            *reinterpret_cast<uint4*>(&sB[buf][row][kv]) =
                *reinterpret_cast<const uint4*>(g);
#endif
        }
    };

    auto compute = [&](int buf) {
        FragB16 a[2], b[2];
#pragma unroll
        for (int ms = 0; ms < 2; ++ms) {
            const u16* r = &sA[buf][wm + ms * 16 + l16][0];
            a[ms].q[0] = *reinterpret_cast<const uint4*>(r + half * 8);
            a[ms].q[1] = *reinterpret_cast<const uint4*>(r + 16 + half * 8);
        }
#pragma unroll
        for (int ns = 0; ns < 2; ++ns) {
            const u16* r = &sB[buf][wn + ns * 16 + l16][0];
            b[ns].q[0] = *reinterpret_cast<const uint4*>(r + half * 8);
            b[ns].q[1] = *reinterpret_cast<const uint4*>(r + 16 + half * 8);
        }
#pragma unroll
        for (int ms = 0; ms < 2; ++ms)
#pragma unroll
            for (int ns = 0; ns < 2; ++ns)
                acc[ms][ns] = wmma_bf16(a[ms], b[ns], acc[ms][ns]);
    };

    const int Kt = K >> 5;
    loadTiles(0, 0);
    async_wait_all();
    __syncthreads();
    for (int kt = 0; kt < Kt; ++kt) {
        int buf = kt & 1;
        if (kt + 1 < Kt) loadTiles(kt + 1, buf ^ 1);   // DMA into other buffer
        compute(buf);                                   // WMMA on current buffer
        async_wait_all();                               // ASYNCcnt -> 0
        __syncthreads();
    }

#pragma unroll
    for (int ms = 0; ms < 2; ++ms)
#pragma unroll
        for (int ns = 0; ns < 2; ++ns) {
            int n = n0 + wn + ns * 16 + l16;
            float bv = bias ? bias[n] : 0.f;
#pragma unroll
            for (int r = 0; r < 8; ++r) {
                int m = bm + wm + ms * 16 + r + half * 8;
                C[(size_t)m * Ncols + n] = acc[ms][ns][r] + bv;
            }
        }
}

// ---------------------------------------------------------------------------
// Predict addressing: per 32-token tile, wave w = head w.  sim = qp_h @ keyn_hᵀ
// via WMMA, softmax(16*sim) entirely in registers (16-lane shfl reductions),
// write bf16 addr[N][8*128] (slots 112..127 zero-padded).
// ---------------------------------------------------------------------------
__global__ __launch_bounds__(256) void predict_addr_kernel(
    const u16* __restrict__ qp,     // [N][512], per-head L2-normalized
    const u16* __restrict__ keyn,   // [8][112][64], row-normalized
    u16* __restrict__ addr)         // [N][1024]
{
    const int tid  = threadIdx.x;
    const int lane = tid & 31, head = tid >> 5;
    const int l16  = lane & 15, half = lane >> 4;
    const int tok0 = blockIdx.x * 32;

    f32x8 acc[2][7];
#pragma unroll
    for (int i = 0; i < 2; ++i)
#pragma unroll
        for (int t = 0; t < 7; ++t) acc[i][t] = zero8();

#pragma unroll
    for (int ks = 0; ks < 2; ++ks) {                 // K = 64 = 2 x 32
        FragB16 a[2];
#pragma unroll
        for (int ms = 0; ms < 2; ++ms) {
            const u16* p = qp + (size_t)(tok0 + ms * 16 + l16) * 512 + head * 64 + ks * 32;
            a[ms].q[0] = *reinterpret_cast<const uint4*>(p + half * 8);
            a[ms].q[1] = *reinterpret_cast<const uint4*>(p + 16 + half * 8);
        }
#pragma unroll
        for (int t = 0; t < 7; ++t) {                // 112 slots = 7 x 16
            const u16* p = keyn + (size_t)(head * 112 + t * 16 + l16) * 64 + ks * 32;
            FragB16 b;
            b.q[0] = *reinterpret_cast<const uint4*>(p + half * 8);
            b.q[1] = *reinterpret_cast<const uint4*>(p + 16 + half * 8);
            acc[0][t] = wmma_bf16(a[0], b, acc[0][t]);
            acc[1][t] = wmma_bf16(a[1], b, acc[1][t]);
        }
    }

#pragma unroll
    for (int ms = 0; ms < 2; ++ms)
#pragma unroll
        for (int r = 0; r < 8; ++r) {
            float mx = -3.0e38f;
#pragma unroll
            for (int t = 0; t < 7; ++t) mx = fmaxf(mx, acc[ms][t][r]);
            mx = wmax16(mx);
            float p[7], s = 0.f;
#pragma unroll
            for (int t = 0; t < 7; ++t) {
                p[t] = __expf(16.f * (acc[ms][t][r] - mx));
                s += p[t];
            }
            s = wsum16(s);
            float inv = 1.f / s;
            int token = tok0 + ms * 16 + r + half * 8;
            u16* o = addr + (size_t)token * 1024 + head * 128;
#pragma unroll
            for (int t = 0; t < 7; ++t) o[t * 16 + l16] = f2bf(p[t] * inv);
            o[112 + l16] = 0;                        // pad slots 112..127
        }
}

// ---------------------------------------------------------------------------
// Recon addressing: wave-private 16-token tile, sim2 = vpn @ value_normᵀ
// (K=512), softmax, write addr2[N][128] (pad zero).
// ---------------------------------------------------------------------------
__global__ __launch_bounds__(256) void recon_addr_kernel(
    const u16* __restrict__ vpn,    // [N][512], row-normalized
    const u16* __restrict__ vnorm,  // [112][512], row-normalized
    u16* __restrict__ addr2)        // [N][128]
{
    const int tid  = threadIdx.x;
    const int lane = tid & 31, w = tid >> 5;
    const int l16  = lane & 15, half = lane >> 4;
    const int tok0 = blockIdx.x * 128 + w * 16;

    f32x8 acc[7];
#pragma unroll
    for (int t = 0; t < 7; ++t) acc[t] = zero8();

    for (int ks = 0; ks < 16; ++ks) {                // K = 512
        FragB16 a;
        const u16* pa = vpn + (size_t)(tok0 + l16) * 512 + ks * 32;
        a.q[0] = *reinterpret_cast<const uint4*>(pa + half * 8);
        a.q[1] = *reinterpret_cast<const uint4*>(pa + 16 + half * 8);
#pragma unroll
        for (int t = 0; t < 7; ++t) {
            const u16* pb = vnorm + (size_t)(t * 16 + l16) * 512 + ks * 32;
            FragB16 b;
            b.q[0] = *reinterpret_cast<const uint4*>(pb + half * 8);
            b.q[1] = *reinterpret_cast<const uint4*>(pb + 16 + half * 8);
            acc[t] = wmma_bf16(a, b, acc[t]);
        }
    }

#pragma unroll
    for (int r = 0; r < 8; ++r) {
        float mx = -3.0e38f;
#pragma unroll
        for (int t = 0; t < 7; ++t) mx = fmaxf(mx, acc[t][r]);
        mx = wmax16(mx);
        float p[7], s = 0.f;
#pragma unroll
        for (int t = 0; t < 7; ++t) {
            p[t] = __expf(16.f * (acc[t][r] - mx));
            s += p[t];
        }
        s = wsum16(s);
        float inv = 1.f / s;
        int token = tok0 + r + half * 8;
        u16* o = addr2 + (size_t)token * 128;
#pragma unroll
        for (int t = 0; t < 7; ++t) o[t * 16 + l16] = f2bf(p[t] * inv);
        o[112 + l16] = 0;
    }
}

// ---------------------------------------------------------------------------
// Row-wise L2 normalize (F.normalize semantics): one wave per row.
// ---------------------------------------------------------------------------
__global__ __launch_bounds__(256) void rownorm_bf16_kernel(
    const float* __restrict__ src, u16* __restrict__ dst,
    float* __restrict__ dstf, int rows, int cols)
{
    int row  = (int)((blockIdx.x * 256 + threadIdx.x) >> 5);
    int lane = threadIdx.x & 31;
    if (row >= rows) return;
    const float* r = src + (size_t)row * cols;
    float s = 0.f;
    for (int c = lane; c < cols; c += 32) { float x = r[c]; s += x * x; }
    s = wsum32(s);
    float inv = 1.f / fmaxf(sqrtf(s), 1e-12f);
    for (int c = lane; c < cols; c += 32) {
        float x = r[c] * inv;
        dst[(size_t)row * cols + c] = f2bf(x);
        if (dstf) dstf[(size_t)row * cols + c] = x;
    }
}

// f_predict = LayerNorm(query + attn) with ln1; one wave per 512-wide row.
__global__ __launch_bounds__(256) void predict_ln_kernel(
    const float* __restrict__ attn, const float* __restrict__ query,
    const float* __restrict__ g, const float* __restrict__ b,
    float* __restrict__ out)
{
    int row  = (int)((blockIdx.x * 256 + threadIdx.x) >> 5);
    int lane = threadIdx.x & 31;
    const float* a = attn + (size_t)row * 512;
    const float* q = query + (size_t)row * 512;
    float x[16], s1 = 0.f, s2 = 0.f;
#pragma unroll
    for (int j = 0; j < 4; ++j) {
        int c = j * 128 + lane * 4;
        float4 av = *reinterpret_cast<const float4*>(a + c);
        float4 qv = *reinterpret_cast<const float4*>(q + c);
        float v0 = av.x + qv.x, v1 = av.y + qv.y, v2 = av.z + qv.z, v3 = av.w + qv.w;
        x[j * 4 + 0] = v0; x[j * 4 + 1] = v1; x[j * 4 + 2] = v2; x[j * 4 + 3] = v3;
        s1 += v0 + v1 + v2 + v3;
        s2 += v0 * v0 + v1 * v1 + v2 * v2 + v3 * v3;
    }
    s1 = wsum32(s1); s2 = wsum32(s2);
    float mean = s1 * (1.f / 512.f);
    float var  = s2 * (1.f / 512.f) - mean * mean;
    float rs   = rsqrtf(var + 1e-5f);
    float* o = out + (size_t)row * 512;
#pragma unroll
    for (int j = 0; j < 4; ++j) {
        int c = j * 128 + lane * 4;
        float4 gv = *reinterpret_cast<const float4*>(g + c);
        float4 bv = *reinterpret_cast<const float4*>(b + c);
        float4 ov;
        ov.x = (x[j * 4 + 0] - mean) * rs * gv.x + bv.x;
        ov.y = (x[j * 4 + 1] - mean) * rs * gv.y + bv.y;
        ov.z = (x[j * 4 + 2] - mean) * rs * gv.z + bv.z;
        ov.w = (x[j * 4 + 3] - mean) * rs * gv.w + bv.w;
        *reinterpret_cast<float4*>(o + c) = ov;
    }
}

// Recon epilogue: cosine recon_loss + LN(ln3) + residual + LN(ln1).
__global__ __launch_bounds__(256) void recon_epilogue_kernel(
    const float* __restrict__ ar, const float* __restrict__ value,
    const float* __restrict__ query,
    const float* __restrict__ g3, const float* __restrict__ b3,
    const float* __restrict__ g1, const float* __restrict__ b1,
    float* __restrict__ out, float* __restrict__ recon_loss)
{
    int row  = (int)((blockIdx.x * 256 + threadIdx.x) >> 5);
    int lane = threadIdx.x & 31;
    const float* ap = ar + (size_t)row * 512;
    const float* vp = value + (size_t)row * 512;
    const float* qp = query + (size_t)row * 512;
    float a[16], v[16], q[16];
    float sa1 = 0.f, sa2 = 0.f, sv2 = 0.f, sdot = 0.f;
#pragma unroll
    for (int j = 0; j < 4; ++j) {
        int c = j * 128 + lane * 4;
        float4 av = *reinterpret_cast<const float4*>(ap + c);
        float4 vv = *reinterpret_cast<const float4*>(vp + c);
        float4 qv = *reinterpret_cast<const float4*>(qp + c);
        a[j*4+0]=av.x; a[j*4+1]=av.y; a[j*4+2]=av.z; a[j*4+3]=av.w;
        v[j*4+0]=vv.x; v[j*4+1]=vv.y; v[j*4+2]=vv.z; v[j*4+3]=vv.w;
        q[j*4+0]=qv.x; q[j*4+1]=qv.y; q[j*4+2]=qv.z; q[j*4+3]=qv.w;
#pragma unroll
        for (int e = 0; e < 4; ++e) {
            float ae = a[j*4+e], ve = v[j*4+e];
            sa1 += ae; sa2 += ae * ae; sv2 += ve * ve; sdot += ae * ve;
        }
    }
    sa1 = wsum32(sa1); sa2 = wsum32(sa2); sv2 = wsum32(sv2); sdot = wsum32(sdot);

    float na  = fmaxf(sqrtf(sa2), 1e-12f);
    float nv  = fmaxf(sqrtf(sv2), 1e-12f);
    float cosv = sdot / (na * nv);
    if (lane == 0)
        atomicAdd(recon_loss, fabsf(1.f - cosv) * (1.f / (float)NTOK));

    // LN3 on attn_recon, then x = query + y, then LN1
    float m3  = sa1 * (1.f / 512.f);
    float v3  = sa2 * (1.f / 512.f) - m3 * m3;
    float rs3 = rsqrtf(v3 + 1e-5f);
    float x[16], s1 = 0.f, s2 = 0.f;
#pragma unroll
    for (int j = 0; j < 4; ++j) {
        int c = j * 128 + lane * 4;
        float4 gv = *reinterpret_cast<const float4*>(g3 + c);
        float4 bv = *reinterpret_cast<const float4*>(b3 + c);
        float y0 = (a[j*4+0]-m3)*rs3*gv.x + bv.x;
        float y1 = (a[j*4+1]-m3)*rs3*gv.y + bv.y;
        float y2 = (a[j*4+2]-m3)*rs3*gv.z + bv.z;
        float y3 = (a[j*4+3]-m3)*rs3*gv.w + bv.w;
        x[j*4+0]=q[j*4+0]+y0; x[j*4+1]=q[j*4+1]+y1;
        x[j*4+2]=q[j*4+2]+y2; x[j*4+3]=q[j*4+3]+y3;
#pragma unroll
        for (int e = 0; e < 4; ++e) { s1 += x[j*4+e]; s2 += x[j*4+e]*x[j*4+e]; }
    }
    s1 = wsum32(s1); s2 = wsum32(s2);
    float mean = s1 * (1.f / 512.f);
    float var  = s2 * (1.f / 512.f) - mean * mean;
    float rs   = rsqrtf(var + 1e-5f);
    float* o = out + (size_t)row * 512;
#pragma unroll
    for (int j = 0; j < 4; ++j) {
        int c = j * 128 + lane * 4;
        float4 gv = *reinterpret_cast<const float4*>(g1 + c);
        float4 bv = *reinterpret_cast<const float4*>(b1 + c);
        float4 ov;
        ov.x = (x[j*4+0]-mean)*rs*gv.x + bv.x;
        ov.y = (x[j*4+1]-mean)*rs*gv.y + bv.y;
        ov.z = (x[j*4+2]-mean)*rs*gv.z + bv.z;
        ov.w = (x[j*4+3]-mean)*rs*gv.w + bv.w;
        *reinterpret_cast<float4*>(o + c) = ov;
    }
}

// ---------------------------------------------------------------------------
// Small prep kernels (all B operands are emitted N-major [Ncols][K] bf16)
// ---------------------------------------------------------------------------
__global__ void convert_bf16_kernel(const float* __restrict__ src,
                                    u16* __restrict__ dst, int n)
{
    int idx = blockIdx.x * 256 + threadIdx.x;
    if (idx < n) dst[idx] = f2bf(src[idx]);
}

// memvalT[d][s] (pad s=112..127 -> 0): B for attn_recon GEMM, [512][128]
__global__ void memval_pad_kernel(const float* __restrict__ mv, u16* __restrict__ dst)
{
    int idx = blockIdx.x * 256 + threadIdx.x;
    if (idx >= 512 * 128) return;
    int d = idx >> 7, s = idx & 127;
    dst[idx] = (s < NSLOT) ? f2bf(mv[(size_t)s * 512 + d]) : (u16)0;
}

// WcatT[o][h*128+s] = sum_c mem_value[s][c] * out_w[o][h*512+c]  (folded GEMM)
__global__ void wcat_kernel(const float* __restrict__ mv,
                            const float* __restrict__ out_w,
                            u16* __restrict__ wcat)
{
    int idx = blockIdx.x * 256 + threadIdx.x;   // [512][1024] outputs
    if (idx >= 512 * 1024) return;
    int o = idx >> 10, p = idx & 1023;
    int h = p >> 7, s = p & 127;
    float acc = 0.f;
    if (s < NSLOT) {
        const float4* mr = reinterpret_cast<const float4*>(mv + (size_t)s * 512);
        const float4* wr = reinterpret_cast<const float4*>(out_w + (size_t)o * 4096 + h * 512);
        for (int c = 0; c < 128; ++c) {
            float4 x = mr[c], y = wr[c];
            acc += x.x * y.x + x.y * y.y + x.z * y.z + x.w * y.w;
        }
    }
    wcat[idx] = f2bf(acc);
}

__global__ void contrastive_kernel(const float* __restrict__ vn, float* __restrict__ cl)
{
    int idx = blockIdx.x * 256 + threadIdx.x;
    if (idx >= NSLOT * NSLOT) return;
    int i = idx / NSLOT, j = idx % NSLOT;
    float d = 0.f;
    const float4* ri = reinterpret_cast<const float4*>(vn + (size_t)i * 512);
    const float4* rj = reinterpret_cast<const float4*>(vn + (size_t)j * 512);
    for (int c = 0; c < 128; ++c) {
        float4 x = ri[c], y = rj[c];
        d += x.x * y.x + x.y * y.y + x.z * y.z + x.w * y.w;
    }
    float t = fabsf(((i == j) ? 1.f : 0.f) - d);
    atomicAdd(cl, 0.01f * t);
}

__global__ void zero2_kernel(float* p) { if (threadIdx.x < 2) p[threadIdx.x] = 0.f; }

// ---------------------------------------------------------------------------
extern "C" void kernel_launch(void* const* d_in, const int* in_sizes, int n_in,
                              void* d_out, int out_size, void* d_ws, size_t ws_size,
                              hipStream_t stream)
{
    const float* query     = (const float*)d_in[0];
    const float* value     = (const float*)d_in[1];
    const float* mem_key   = (const float*)d_in[2];
    const float* mem_value = (const float*)d_in[3];
    const float* q_w   = (const float*)d_in[4];
    const float* q_b   = (const float*)d_in[5];
    const float* v_w   = (const float*)d_in[6];
    const float* v_b   = (const float*)d_in[7];
    const float* out_w = (const float*)d_in[8];
    const float* out_b = (const float*)d_in[9];
    const float* ln1_g = (const float*)d_in[10];
    const float* ln1_b = (const float*)d_in[11];
    const float* ln3_g = (const float*)d_in[12];
    const float* ln3_b = (const float*)d_in[13];

    char* ws = (char*)d_ws;
    size_t off = 0;
    auto alloc = [&](size_t bytes) -> void* {
        void* p = ws + off;
        off = (off + bytes + 255) & ~(size_t)255;
        return p;
    };
    float* f32buf   = (float*)alloc((size_t)NTOK * 512 * 4);   // qp/attn/vproj/ar (reused)
    u16*   bf1      = (u16*)  alloc((size_t)NTOK * 512 * 2);   // qp_bf16 / vpn_bf16
    u16*   bf2      = (u16*)  alloc((size_t)NTOK * 1024 * 2);  // addr / addr2
    u16*   keyn     = (u16*)  alloc((size_t)NHEAD * NSLOT * HDIM * 2);
    u16*   vnorm_bf = (u16*)  alloc((size_t)NSLOT * 512 * 2);
    float* vnorm_f  = (float*)alloc((size_t)NSLOT * 512 * 4);
    u16*   memvalT  = (u16*)  alloc((size_t)512 * 128 * 2);    // [512][128]
    u16*   qwB      = (u16*)  alloc((size_t)512 * 512 * 2);    // [out][in] = [N][K]
    u16*   vwB      = (u16*)  alloc((size_t)512 * 512 * 2);
    u16*   wcatT    = (u16*)  alloc((size_t)512 * 1024 * 2);   // [512][1024]

    float* f_pred = (float*)d_out;
    float* f_tgt  = f_pred + (size_t)NTOK * 512;
    float* losses = f_pred + (size_t)2 * NTOK * 512;   // [recon, contrastive]

    // ---- prep ----
    zero2_kernel<<<1, 32, 0, stream>>>(losses);
    rownorm_bf16_kernel<<<(NHEAD * NSLOT + 7) / 8, 256, 0, stream>>>(
        mem_key, keyn, nullptr, NHEAD * NSLOT, HDIM);
    rownorm_bf16_kernel<<<(NSLOT + 7) / 8, 256, 0, stream>>>(
        mem_value, vnorm_bf, vnorm_f, NSLOT, 512);
    convert_bf16_kernel<<<(512 * 512) / 256, 256, 0, stream>>>(q_w, qwB, 512 * 512);
    convert_bf16_kernel<<<(512 * 512) / 256, 256, 0, stream>>>(v_w, vwB, 512 * 512);
    memval_pad_kernel<<<(512 * 128) / 256, 256, 0, stream>>>(mem_value, memvalT);
    wcat_kernel<<<(512 * 1024) / 256, 256, 0, stream>>>(mem_value, out_w, wcatT);
    contrastive_kernel<<<(NSLOT * NSLOT + 255) / 256, 256, 0, stream>>>(
        vnorm_f, losses + 1);

    // ---- predict branch ----
    dim3 gemm_grid(512 / 64, NTOK / 128);
    gemm_bias_kernel<float><<<gemm_grid, 256, 0, stream>>>(
        query, qwB, q_b, f32buf, NTOK, 512, 512);                 // qp
    rownorm_bf16_kernel<<<(NTOK * NHEAD) / 8, 256, 0, stream>>>(
        f32buf, bf1, nullptr, NTOK * NHEAD, HDIM);                // per-head L2
    predict_addr_kernel<<<NTOK / 32, 256, 0, stream>>>(bf1, keyn, bf2);
    gemm_bias_kernel<u16><<<gemm_grid, 256, 0, stream>>>(
        bf2, wcatT, out_b, f32buf, NTOK, 512, 1024);              // attn_out
    predict_ln_kernel<<<NTOK / 8, 256, 0, stream>>>(
        f32buf, query, ln1_g, ln1_b, f_pred);

    // ---- recon branch ----
    gemm_bias_kernel<float><<<gemm_grid, 256, 0, stream>>>(
        value, vwB, v_b, f32buf, NTOK, 512, 512);                 // v_proj
    rownorm_bf16_kernel<<<NTOK / 8, 256, 0, stream>>>(
        f32buf, bf1, nullptr, NTOK, 512);                         // row L2
    recon_addr_kernel<<<NTOK / 128, 256, 0, stream>>>(bf1, vnorm_bf, bf2);
    gemm_bias_kernel<u16><<<gemm_grid, 256, 0, stream>>>(
        bf2, memvalT, nullptr, f32buf, NTOK, 512, 128);           // attn_recon
    recon_epilogue_kernel<<<NTOK / 8, 256, 0, stream>>>(
        f32buf, value, query, ln3_g, ln3_b, ln1_g, ln1_b, f_tgt, losses);
}
